// Hierarchical12GNN_10797547782338
// MI455X (gfx1250) — compile-verified, matrix operand order checked by reading, not run on hardware
//
#include <hip/hip_runtime.h>
#include <hip/hip_bf16.h>

// ---------------------------------------------------------------------------
// Types for CDNA5 WMMA (wave32): v_wmma_f32_16x16x32_bf16
// ---------------------------------------------------------------------------
typedef __attribute__((ext_vector_type(16))) __bf16 v16bf;
typedef __attribute__((ext_vector_type(8)))  float  v8f;
typedef __attribute__((ext_vector_type(4)))  unsigned int u32x4;

union AFrag { u32x4 u[2]; v16bf v; };

// Load one 16x32 bf16 WMMA operand fragment for this lane.
// Per ISA layout: lane holds M(or N) = lane&15; VGPRs 0..3 hold
// K = half*8 + [0..7] (contiguous!), VGPRs 4..7 hold K = 16 + half*8 + [0..7].
// => exactly two 16-byte loads per fragment.
__device__ __forceinline__ v16bf load_frag(const __bf16* __restrict__ base,
                                           int ld, int row, int k0, int half) {
    AFrag f;
    const __bf16* p = base + (size_t)row * ld + k0 + half * 8;
    f.u[0] = *reinterpret_cast<const u32x4*>(p);
    f.u[1] = *reinterpret_cast<const u32x4*>(p + 16);
    return f.v;
}

// ---------------------------------------------------------------------------
// C[M x 128] = A[M x K](bf16, row-major) @ W[128 x K](bf16, row-major)^T
// Block: 128 threads = 4 waves; wave w computes a 16x128 output strip
// (rows [bx*64 + 16w, +16), ALL 8 column tiles): each A fragment is loaded
// once per K-step and reused across 8 WMMAs. All 8 B fragments of a K-step
// are materialized in distinct registers BEFORE the WMMA chain so loads can
// issue as one clause and overlap WMMA execution (incremental loadcnt waits).
// M % 64 == 0, K % 32 == 0, N == 128 exactly: no divergence -> EXEC all 1s.
// out_bf16: 0 -> write float, 1 -> write bf16
// ---------------------------------------------------------------------------
__global__ void __launch_bounds__(128)
gemm_bf16_wmma(const __bf16* __restrict__ A, const __bf16* __restrict__ W,
               void* __restrict__ C, int N, int K, int out_bf16) {
    const int wave = threadIdx.x >> 5;
    const int lane = threadIdx.x & 31;
    const int half = lane >> 4;
    const int l16  = lane & 15;
    const int row_base = blockIdx.x * 64 + wave * 16;
    const int arow = row_base + l16;

    v8f zero = {};
    v8f acc[8];
    #pragma unroll
    for (int c = 0; c < 8; ++c) acc[c] = zero;

    for (int k0 = 0; k0 < K; k0 += 32) {
        v16bf a = load_frag(A, K, arow, k0, half);
        v16bf b[8];
        #pragma unroll
        for (int c = 0; c < 8; ++c)
            b[c] = load_frag(W, K, c * 16 + l16, k0, half);
        #pragma unroll
        for (int c = 0; c < 8; ++c)
            acc[c] = __builtin_amdgcn_wmma_f32_16x16x32_bf16(
                false, a, false, b[c], (short)0, acc[c], false, false);
    }
    // D layout: VGPR r -> M = r + 8*half ; N = lane&15
    if (out_bf16) {
        __bf16* out = (__bf16*)C;
        #pragma unroll
        for (int c = 0; c < 8; ++c) {
            #pragma unroll
            for (int r = 0; r < 8; ++r) {
                int m = row_base + r + half * 8;
                out[(size_t)m * N + c * 16 + l16] = (__bf16)acc[c][r];
            }
        }
    } else {
        float* out = (float*)C;
        #pragma unroll
        for (int c = 0; c < 8; ++c) {
            #pragma unroll
            for (int r = 0; r < 8; ++r) {
                int m = row_base + r + half * 8;
                out[(size_t)m * N + c * 16 + l16] = acc[c][r];
            }
        }
    }
}

// ---------------------------------------------------------------------------
// Elementwise helpers
// ---------------------------------------------------------------------------
__global__ void cvt_f32_bf16(const float* __restrict__ in,
                             __bf16* __restrict__ out, long n) {
    long i = (long)blockIdx.x * 256 + threadIdx.x;
    if (i < n) out[i] = (__bf16)in[i];
}

__global__ void relu_cvt_bf16(const float* __restrict__ in,
                              __bf16* __restrict__ out, long n) {
    long i = (long)blockIdx.x * 256 + threadIdx.x;
    if (i < n) out[i] = (__bf16)fmaxf(in[i], 0.0f);
}

// Split k1 weight [128 x 257] into bf16 [:, :128], [:,128:256] and f32 col 256
__global__ void split_k1(const float* __restrict__ k1w1, const float* __restrict__ k1w2,
                         __bf16* __restrict__ WA, __bf16* __restrict__ WB,
                         __bf16* __restrict__ WC, __bf16* __restrict__ WD,
                         float* __restrict__ w1c, float* __restrict__ w2c) {
    int i = blockIdx.x * 128 + threadIdx.x;     // 16384 total
    int n = i >> 7, k = i & 127;
    WA[i] = (__bf16)k1w1[n * 257 + k];
    WB[i] = (__bf16)k1w1[n * 257 + 128 + k];
    WC[i] = (__bf16)k1w2[n * 257 + k];
    WD[i] = (__bf16)k1w2[n * 257 + 128 + k];
    if (k == 0) { w1c[n] = k1w1[n * 257 + 256]; w2c[n] = k1w2[n * 257 + 256]; }
}

// ---------------------------------------------------------------------------
// Edge scatter: acc[dst[e], :] += msg[src[e], :]   (bf16 msg, f32 atomic acc)
// One wave per edge, 4 dims per lane.
// ---------------------------------------------------------------------------
__global__ void edge_scatter(const __bf16* __restrict__ msg,
                             const int* __restrict__ src, const int* __restrict__ dst,
                             float* __restrict__ acc, int E) {
    int e = blockIdx.x * 4 + (threadIdx.x >> 5);
    if (e >= E) return;
    int lane = threadIdx.x & 31;
    const __bf16* mp = msg + (size_t)src[e] * 128;
    float*        ap = acc + (size_t)dst[e] * 128;
    #pragma unroll
    for (int i = 0; i < 4; ++i) {
        int d = lane + i * 32;
        atomicAdd(&ap[d], (float)mp[d]);
    }
}

// ---------------------------------------------------------------------------
// Pair-level assembly (the distributive-trick fusion of the k1 layer):
//   acc[p,d] = HA[pu[p],d] + HB[pv[p],d] + iso[p]*w1c[d]        (self part, f32)
//   msg[p,d] = bf16( HC[pu[p],d] + HD[pv[p],d] + iso[p]*w2c[d] ) (message part)
// ---------------------------------------------------------------------------
__global__ void pair_assemble(const float* __restrict__ HA, const float* __restrict__ HB,
                              const float* __restrict__ HC, const float* __restrict__ HD,
                              const float* __restrict__ w1c, const float* __restrict__ w2c,
                              const float* __restrict__ iso,
                              const int* __restrict__ pu, const int* __restrict__ pv,
                              float* __restrict__ acc, __bf16* __restrict__ msg) {
    int p = blockIdx.x, d = threadIdx.x;
    int u = pu[p], v = pv[p];
    float is = iso[p];
    size_t o = (size_t)p * 128 + d;
    acc[o] = HA[(size_t)u * 128 + d] + HB[(size_t)v * 128 + d] + is * w1c[d];
    msg[o] = (__bf16)(HC[(size_t)u * 128 + d] + HD[(size_t)v * 128 + d] + is * w2c[d]);
}

// emb1[g,d] = sum over 64 nodes of h(bf16)
__global__ void seg_sum_nodes(const __bf16* __restrict__ h, float* __restrict__ emb1) {
    int g = blockIdx.x, d = threadIdx.x;
    const __bf16* p = h + (size_t)g * 64 * 128 + d;
    float s = 0.f;
    for (int v = 0; v < 64; ++v) s += (float)p[(size_t)v * 128];
    emb1[g * 128 + d] = s;
}

// emb2[g,d] = sum over P pairs of relu(acc)
__global__ void seg_sum_pairs(const float* __restrict__ acc, float* __restrict__ emb2, int P) {
    int g = blockIdx.x, d = threadIdx.x;
    const float* p = acc + (size_t)g * P * 128 + d;
    float s = 0.f;
    for (int q = 0; q < P; ++q) s += fmaxf(p[(size_t)q * 128], 0.f);
    emb2[g * 128 + d] = s;
}

// Classifier: out[g] = relu([emb1,emb2] @ W1^T + b1) @ W2^T + b2
__global__ void classifier(const float* __restrict__ emb1, const float* __restrict__ emb2,
                           const float* __restrict__ w1, const float* __restrict__ b1,
                           const float* __restrict__ w2, const float* __restrict__ b2,
                           float* __restrict__ out) {
    __shared__ float hid[128];
    int g = blockIdx.x, t = threadIdx.x;
    const float* e1 = emb1 + g * 128;
    const float* e2 = emb2 + g * 128;
    const float* wr = w1 + t * 256;
    float s = b1[t];
    for (int j = 0; j < 128; ++j) s += e1[j] * wr[j];
    for (int j = 0; j < 128; ++j) s += e2[j] * wr[128 + j];
    hid[t] = fmaxf(s, 0.f);
    __syncthreads();
    if (t < 10) {
        float o = b2[t];
        const float* w2r = w2 + t * 128;
        for (int j = 0; j < 128; ++j) o += hid[j] * w2r[j];
        out[g * 10 + t] = o;
    }
}

// ---------------------------------------------------------------------------
// Host-side orchestration
// ---------------------------------------------------------------------------
extern "C" void kernel_launch(void* const* d_in, const int* in_sizes, int n_in,
                              void* d_out, int out_size, void* d_ws, size_t ws_size,
                              hipStream_t stream) {
    (void)n_in; (void)out_size; (void)ws_size;
    const float* x      = (const float*)d_in[0];
    const float* g1_w1  = (const float*)d_in[1];
    const float* g1_w2  = (const float*)d_in[2];
    const float* k1_w1  = (const float*)d_in[3];
    const float* k1_w2  = (const float*)d_in[4];
    const float* k2_w1  = (const float*)d_in[5];
    const float* k2_w2  = (const float*)d_in[6];
    const float* cls_w1 = (const float*)d_in[7];
    const float* cls_b1 = (const float*)d_in[8];
    const float* cls_w2 = (const float*)d_in[9];
    const float* cls_b2 = (const float*)d_in[10];
    const float* iso    = (const float*)d_in[11];
    const int* ei       = (const int*)d_in[12];
    const int* pu       = (const int*)d_in[14];
    const int* pv       = (const int*)d_in[15];
    const int* tei      = (const int*)d_in[17];
    float* out = (float*)d_out;

    const int NNODES = in_sizes[0] / 128;        // 8192
    const int E1     = in_sizes[12] / 2;
    const int PTOT   = in_sizes[14];             // 258048
    const int E2     = in_sizes[17] / 2;
    const int B      = 128;
    const int PPG    = PTOT / B;                 // 2016
    const int HID    = 128;

    // ---- workspace carve-up -------------------------------------------------
    size_t off = 0;
    auto carve = [&](size_t bytes) {
        void* p = (char*)d_ws + off;
        off += (bytes + 255) & ~(size_t)255;
        return p;
    };
    __bf16* HBF_A  = (__bf16*)carve((size_t)NNODES * HID * 2);
    __bf16* HBF_B  = (__bf16*)carve((size_t)NNODES * HID * 2);
    float*  Y1     = (float*) carve((size_t)NNODES * HID * 4);
    __bf16* Y2BF   = (__bf16*)carve((size_t)NNODES * HID * 2);
    __bf16* G1W1BF = (__bf16*)carve((size_t)3 * HID * HID * 2);
    __bf16* G1W2BF = (__bf16*)carve((size_t)3 * HID * HID * 2);
    __bf16* WA     = (__bf16*)carve((size_t)HID * HID * 2);
    __bf16* WB     = (__bf16*)carve((size_t)HID * HID * 2);
    __bf16* WC     = (__bf16*)carve((size_t)HID * HID * 2);
    __bf16* WD     = (__bf16*)carve((size_t)HID * HID * 2);
    __bf16* K2W1BF = (__bf16*)carve((size_t)HID * HID * 2);
    __bf16* K2W2BF = (__bf16*)carve((size_t)HID * HID * 2);
    float*  W1C    = (float*) carve(HID * 4);
    float*  W2C    = (float*) carve(HID * 4);
    float*  HA     = (float*) carve((size_t)NNODES * HID * 4);
    float*  HB     = (float*) carve((size_t)NNODES * HID * 4);
    float*  HC     = (float*) carve((size_t)NNODES * HID * 4);
    float*  HD     = (float*) carve((size_t)NNODES * HID * 4);
    float*  EMB1   = (float*) carve((size_t)B * HID * 4);
    float*  EMB2   = (float*) carve((size_t)B * HID * 4);
    float*  PACC   = (float*) carve((size_t)PTOT * HID * 4);   // 132 MB
    __bf16* PMSG   = (__bf16*)carve((size_t)PTOT * HID * 2);   //  66 MB
    __bf16* H2BF   = (__bf16*)carve((size_t)PTOT * HID * 2);   //  66 MB

    // ---- convert weights to bf16 -------------------------------------------
    long nW1 = 3L * HID * HID;
    cvt_f32_bf16<<<(nW1 + 255) / 256, 256, 0, stream>>>(g1_w1, G1W1BF, nW1);
    cvt_f32_bf16<<<(nW1 + 255) / 256, 256, 0, stream>>>(g1_w2, G1W2BF, nW1);
    long nW = (long)HID * HID;
    cvt_f32_bf16<<<(nW + 255) / 256, 256, 0, stream>>>(k2_w1, K2W1BF, nW);
    cvt_f32_bf16<<<(nW + 255) / 256, 256, 0, stream>>>(k2_w2, K2W2BF, nW);
    split_k1<<<128, 128, 0, stream>>>(k1_w1, k1_w2, WA, WB, WC, WD, W1C, W2C);

    // ---- node stage: 3 message-passing layers ------------------------------
    long nH = (long)NNODES * HID;
    cvt_f32_bf16<<<(nH + 255) / 256, 256, 0, stream>>>(x, HBF_A, nH);

    dim3 gNode(NNODES / 64);
    __bf16* hin = HBF_A; __bf16* hout = HBF_B;
    for (int layer = 0; layer < 3; ++layer) {
        const __bf16* w1 = G1W1BF + (size_t)layer * HID * HID;
        const __bf16* w2 = G1W2BF + (size_t)layer * HID * HID;
        gemm_bf16_wmma<<<gNode, 128, 0, stream>>>(hin, w1, Y1,   HID, HID, 0);
        gemm_bf16_wmma<<<gNode, 128, 0, stream>>>(hin, w2, Y2BF, HID, HID, 1);
        edge_scatter<<<(E1 + 3) / 4, 128, 0, stream>>>(Y2BF, ei, ei + E1, Y1, E1);
        relu_cvt_bf16<<<(nH + 255) / 256, 256, 0, stream>>>(Y1, hout, nH);
        __bf16* t = hin; hin = hout; hout = t;
    }
    // hin now holds the final node embeddings h (bf16)
    seg_sum_nodes<<<B, 128, 0, stream>>>(hin, EMB1);

    // ---- k1 layer via distributive trick -----------------------------------
    gemm_bf16_wmma<<<gNode, 128, 0, stream>>>(hin, WA, HA, HID, HID, 0);
    gemm_bf16_wmma<<<gNode, 128, 0, stream>>>(hin, WB, HB, HID, HID, 0);
    gemm_bf16_wmma<<<gNode, 128, 0, stream>>>(hin, WC, HC, HID, HID, 0);
    gemm_bf16_wmma<<<gNode, 128, 0, stream>>>(hin, WD, HD, HID, HID, 0);
    pair_assemble<<<PTOT, 128, 0, stream>>>(HA, HB, HC, HD, W1C, W2C, iso,
                                            pu, pv, PACC, PMSG);
    edge_scatter<<<(E2 + 3) / 4, 128, 0, stream>>>(PMSG, tei, tei + E2, PACC, E2);

    long nP = (long)PTOT * HID;
    relu_cvt_bf16<<<(nP + 255) / 256, 256, 0, stream>>>(PACC, H2BF, nP);

    // ---- k2 layer: full-size GEMMs on pair activations ---------------------
    dim3 gPair(PTOT / 64);
    gemm_bf16_wmma<<<gPair, 128, 0, stream>>>(H2BF, K2W1BF, PACC, HID, HID, 0);
    gemm_bf16_wmma<<<gPair, 128, 0, stream>>>(H2BF, K2W2BF, PMSG, HID, HID, 1);
    edge_scatter<<<(E2 + 3) / 4, 128, 0, stream>>>(PMSG, tei, tei + E2, PACC, E2);

    // relu fused into the segment sum
    seg_sum_pairs<<<B, 128, 0, stream>>>(PACC, EMB2, PPG);

    // ---- classifier ---------------------------------------------------------
    classifier<<<B, 128, 0, stream>>>(EMB1, EMB2, cls_w1, cls_b1, cls_w2, cls_b2, out);
}
